// GroupAgent_72945724555759
// MI455X (gfx1250) — matrix operand order repaired and checked by reading, not compiled, not run on hardware
//
#include <hip/hip_runtime.h>
#include <hip/hip_bf16.h>
#include <math.h>

#define B_   2048
#define A_   32
#define EIN  160
#define H_   128
#define NACT 32
#define HE_  256
#define G_   3

typedef __bf16 bf16;
typedef __attribute__((ext_vector_type(16))) __bf16 v16bf;
typedef __attribute__((ext_vector_type(8)))  float  v8f;

__device__ __forceinline__ bf16 f2bf(float f) { return (bf16)f; }

__device__ __forceinline__ v8f wmma_bf16(v16bf a, v16bf b, v8f c) {
  return __builtin_amdgcn_wmma_f32_16x16x32_bf16(false, a, false, b, (short)0, c,
                                                 false, false);
}

// 16x32 A fragment from row-major f32 (LDS or global), contiguous K order:
// lanes 0-15 hold K {kb..kb+15}; lanes 16-31 hold {kb+16..kb+31}
__device__ __forceinline__ v16bf load_a(const float* A, int ld, int kbase) {
  int lane = threadIdx.x & 31;
  int row  = lane & 15;
  int kh   = (lane >> 4) << 4;           // 0 or 16
  const float* r = A + (size_t)row * ld + kbase;
  v16bf a;
#pragma unroll
  for (int i = 0; i < 8; ++i) a[i]     = f2bf(r[kh + i]);
#pragma unroll
  for (int i = 0; i < 8; ++i) a[i + 8] = f2bf(r[8 + kh + i]);
  return a;
}

// pre-packed 32x16 B fragment: 512 bf16 per tile, lane-contiguous 32B per lane
__device__ __forceinline__ v16bf load_pb(const bf16* packed, int tile) {
  const v16bf* p = (const v16bf*)(packed + (size_t)tile * 512);
  return p[threadIdx.x & 31];
}

// store 16x16 C tile (f32 C/D layout: VGPR j -> M=j or j+8; N=lane&15)
__device__ __forceinline__ void store_c(float* C, int ld, v8f c) {
  int lane = threadIdx.x & 31;
  int col  = lane & 15;
  int rb   = (lane >> 4) << 3;
#pragma unroll
  for (int j = 0; j < 8; ++j) C[(size_t)(rb + j) * ld + col] = c[j];
}

// ---------------------------------------------------------------------------
// Weight packer: src [K x N] row-major f32 -> B-fragment bf16 tiles.
// dst[((ct*KC+kc)*32 + lane)*16 + i] = src[kc*32 + ((lane>>4)<<4) + i][ct*16 + (lane&15)]
// zero-padded for rows >= K (handles se1_w K=34 -> 64).
// ---------------------------------------------------------------------------
__global__ void pack_b_kernel(const float* __restrict__ src, bf16* __restrict__ dst,
                              int K, int N) {
  int KC = (K + 31) >> 5;
  int NT = N >> 4;
  int total = NT * KC * 512;
  int idx = blockIdx.x * blockDim.x + threadIdx.x;
  if (idx >= total) return;
  int i    = idx & 15;
  int lane = (idx >> 4) & 31;
  int tile = idx >> 9;
  int kc   = tile % KC;
  int ct   = tile / KC;
  int row  = kc * 32 + ((lane >> 4) << 4) + i;
  int col  = ct * 16 + (lane & 15);
  float v = (row < K) ? src[(size_t)row * N + col] : 0.0f;
  dst[idx] = f2bf(v);
}

// ---------------------------------------------------------------------------
// Kernel 1: x = relu(inp@fc1 + b); GRU cell -> h_out [R,128]
// 4 waves/block, one 16-row tile per wave.
// ---------------------------------------------------------------------------
__global__ void __launch_bounds__(128)
k_fc1_gru(const float* __restrict__ inp, const float* __restrict__ h_in,
          const float* __restrict__ fc1_b, const float* __restrict__ gbi,
          const float* __restrict__ gbh,
          const bf16* __restrict__ pw_fc1, const bf16* __restrict__ pw_wi,
          const bf16* __restrict__ pw_wh, float* __restrict__ h_out) {
  __shared__ float lds_x[4][16 * 128];
  int wid = threadIdx.x >> 5;
  int lane = threadIdx.x & 31;
  int rowbase = (blockIdx.x * 4 + wid) * 16;
  float* xs = lds_x[wid];
  int col = lane & 15;
  int rb  = (lane >> 4) << 3;

  // ---- fc1: [16,160] @ [160,128], K chunks = 5 ----
  v16bf afr[5];
#pragma unroll
  for (int kc = 0; kc < 5; ++kc)
    afr[kc] = load_a(inp + (size_t)rowbase * EIN, EIN, kc * 32);
#pragma unroll
  for (int nt = 0; nt < 8; ++nt) {
    v8f acc = {};
#pragma unroll
    for (int kc = 0; kc < 5; ++kc)
      acc = wmma_bf16(afr[kc], load_pb(pw_fc1, nt * 5 + kc), acc);
    float b = fc1_b[nt * 16 + col];
#pragma unroll
    for (int j = 0; j < 8; ++j) { float v = acc[j] + b; acc[j] = v > 0.f ? v : 0.f; }
    store_c(xs + nt * 16, 128, acc);
  }

  // ---- GRU: gi = x@Wi, gh = h_in@Wh (K=128 each) ----
  v16bf ax[4], ah[4];
#pragma unroll
  for (int kc = 0; kc < 4; ++kc) {
    ax[kc] = load_a(xs, 128, kc * 32);
    ah[kc] = load_a(h_in + (size_t)rowbase * H_, H_, kc * 32);
  }
#pragma unroll 1
  for (int nt = 0; nt < 8; ++nt) {
    v8f ir = {}, iz = {}, inn = {}, hr = {}, hz = {}, hn = {};
#pragma unroll
    for (int kc = 0; kc < 4; ++kc) {
      ir  = wmma_bf16(ax[kc], load_pb(pw_wi, (nt)      * 4 + kc), ir);
      iz  = wmma_bf16(ax[kc], load_pb(pw_wi, (nt + 8)  * 4 + kc), iz);
      inn = wmma_bf16(ax[kc], load_pb(pw_wi, (nt + 16) * 4 + kc), inn);
      hr  = wmma_bf16(ah[kc], load_pb(pw_wh, (nt)      * 4 + kc), hr);
      hz  = wmma_bf16(ah[kc], load_pb(pw_wh, (nt + 8)  * 4 + kc), hz);
      hn  = wmma_bf16(ah[kc], load_pb(pw_wh, (nt + 16) * 4 + kc), hn);
    }
    int gcol = nt * 16 + col;
    float bir = gbi[gcol], biz = gbi[128 + gcol], bin = gbi[256 + gcol];
    float bhr = gbh[gcol], bhz = gbh[128 + gcol], bhn = gbh[256 + gcol];
#pragma unroll
    for (int j = 0; j < 8; ++j) {
      size_t row = (size_t)(rowbase + rb + j);
      float r = 1.f / (1.f + __expf(-(ir[j] + bir + hr[j] + bhr)));
      float z = 1.f / (1.f + __expf(-(iz[j] + biz + hz[j] + bhz)));
      float n = tanhf(inn[j] + bin + r * (hn[j] + bhn));
      float hv = h_in[row * H_ + gcol];
      h_out[row * H_ + gcol] = (1.f - z) * n + z * hv;
    }
  }
}

// ---------------------------------------------------------------------------
// Kernel 2: attention per batch. Block = 1 batch, 128 threads.
// Outputs attn [32,32] to d_out and struct_input [32,64] (zero-padded) to ws.
// ---------------------------------------------------------------------------
__global__ void __launch_bounds__(128)
k_attn(const float* __restrict__ h, const bf16* __restrict__ pw_q,
       const bf16* __restrict__ pw_k, float* __restrict__ attn_out,
       float* __restrict__ si_out) {
  __shared__ float hs[32 * 128];
  __shared__ float qm[32 * 128];
  __shared__ float km[32 * 128];
  __shared__ float sc[32 * 32];
  __shared__ float at[32 * 32];
  int b = blockIdx.x;
  int tid = threadIdx.x;
  const float* hb = h + (size_t)b * 32 * 128;
  for (int i = tid; i < 32 * 128; i += 128) hs[i] = hb[i];
  __syncthreads();

  int wid = tid >> 5;
  int lane = tid & 31;

  // q_ = h@Wq, k_ = h@Wk : 32 tiles, 8 per wave
  for (int t = wid; t < 32; t += 4) {
    int mat = t >> 4;              // 0=q, 1=k
    int rt  = (t >> 3) & 1;
    int ct  = t & 7;
    v8f acc = {};
#pragma unroll
    for (int kc = 0; kc < 4; ++kc) {
      v16bf a = load_a(hs + rt * 16 * 128, 128, kc * 32);
      acc = wmma_bf16(a, load_pb(mat ? pw_k : pw_q, ct * 4 + kc), acc);
    }
    store_c((mat ? km : qm) + rt * 16 * 128 + ct * 16, 128, acc);
  }
  __syncthreads();

  // scores = q_ @ k_^T / sqrt(H): 4 tiles, one per wave
  {
    int rt = wid >> 1, ct = wid & 1;
    int coln = ct * 16 + (lane & 15);
    int koff = (lane >> 4) << 4;
    v8f acc = {};
#pragma unroll
    for (int kc = 0; kc < 4; ++kc) {
      v16bf a = load_a(qm + rt * 16 * 128, 128, kc * 32);
      v16bf bf;
#pragma unroll
      for (int i = 0; i < 16; ++i)
        bf[i] = f2bf(km[coln * 128 + kc * 32 + koff + i]);
      acc = wmma_bf16(a, bf, acc);
    }
    const float scale = 0.08838834764831845f;   // 1/sqrt(128)
#pragma unroll
    for (int j = 0; j < 8; ++j) acc[j] *= scale;
    store_c(sc + rt * 16 * 32 + ct * 16, 32, acc);
  }
  __syncthreads();

  // masked softmax per row
  if (tid < 32) {
    int r = tid;
    sc[r * 32 + r] = -1e9f;
    float m = -1e30f;
    for (int j = 0; j < 32; ++j) m = fmaxf(m, sc[r * 32 + j]);
    float s = 0.f;
    for (int j = 0; j < 32; ++j) {
      float e = __expf(sc[r * 32 + j] - m);
      at[r * 32 + j] = e; s += e;
    }
    float inv = 1.f / s;
    for (int j = 0; j < 32; ++j) at[r * 32 + j] *= inv;
  }
  __syncthreads();

  // symmetrize, zero diag, degree + entropy, emit outputs
  if (tid < 32) {
    int r = tid;
    float deg = 0.f, ent = 0.f;
    float* si = si_out + ((size_t)b * 32 + r) * 64;
    float* ao = attn_out + ((size_t)b * 32 + r) * 32;
    for (int j = 0; j < 32; ++j) {
      float v = (j == r) ? 0.f : 0.5f * (at[r * 32 + j] + at[j * 32 + r]);
      ao[j] = v; si[j] = v;
      deg += v;
      float vc = fmaxf(v, 1e-8f);
      ent -= vc * __logf(vc);
    }
    si[32] = deg; si[33] = ent;
    for (int j = 34; j < 64; ++j) si[j] = 0.f;
  }
}

// ---------------------------------------------------------------------------
// Kernel 3: struct encoder -> probs -> group decoder -> fused hypernetwork.
// One wave per block, one 16-row tile per wave.
// ---------------------------------------------------------------------------
__global__ void __launch_bounds__(32)
k_group(const float* __restrict__ si, const float* __restrict__ h,
        const float* __restrict__ se1_b, const float* __restrict__ se2_b,
        const float* __restrict__ ga_w, const float* __restrict__ ga_b,
        const float* __restrict__ gemb, const float* __restrict__ gd1_b,
        const float* __restrict__ gd2_b, const float* __restrict__ hb_b,
        const float* __restrict__ hw_b,
        const bf16* __restrict__ pw_se1, const bf16* __restrict__ pw_se2,
        const bf16* __restrict__ pw_gd1, const bf16* __restrict__ pw_gd2,
        const bf16* __restrict__ pw_hb,  const bf16* __restrict__ pw_hw,
        float* __restrict__ q_out, float* __restrict__ gs_out,
        float* __restrict__ probs_out) {
  __shared__ float bufA[16 * 256];
  __shared__ float bufB[16 * 256];
  __shared__ float hloc[16 * 128];
  int lane = threadIdx.x;
  int rowbase = blockIdx.x * 16;
  int col = lane & 15;
  int rb  = (lane >> 4) << 3;

  for (int i = lane; i < 16 * 128; i += 32)
    hloc[i] = h[(size_t)rowbase * 128 + i];

  // se1: relu([16,64]@[64,256] + b) -> bufA
  {
    v16bf a0 = load_a(si + (size_t)rowbase * 64, 64, 0);
    v16bf a1 = load_a(si + (size_t)rowbase * 64, 64, 32);
#pragma unroll 1
    for (int ct = 0; ct < 16; ++ct) {
      v8f acc = {};
      acc = wmma_bf16(a0, load_pb(pw_se1, ct * 2 + 0), acc);
      acc = wmma_bf16(a1, load_pb(pw_se1, ct * 2 + 1), acc);
      float bb = se1_b[ct * 16 + col];
#pragma unroll
      for (int j = 0; j < 8; ++j) { float v = acc[j] + bb; acc[j] = v > 0.f ? v : 0.f; }
      store_c(bufA + ct * 16, 256, acc);
    }
  }
  // sf = tanh(bufA@se2 + b) -> bufB
  {
    v16bf afr[8];
#pragma unroll
    for (int kc = 0; kc < 8; ++kc) afr[kc] = load_a(bufA, 256, kc * 32);
#pragma unroll 1
    for (int ct = 0; ct < 16; ++ct) {
      v8f acc = {};
#pragma unroll
      for (int kc = 0; kc < 8; ++kc)
        acc = wmma_bf16(afr[kc], load_pb(pw_se2, ct * 8 + kc), acc);
      float bb = se2_b[ct * 16 + col];
#pragma unroll
      for (int j = 0; j < 8; ++j) acc[j] = tanhf(acc[j] + bb);
      store_c(bufB + ct * 16, 256, acc);
    }
  }
  // probs (softmax over G=3) + gd_in = sf + probs@gemb -> bufA
  {
    int r = lane & 15;
    const float* sfr = bufB + r * 256;
    float lg[3];
#pragma unroll
    for (int g = 0; g < 3; ++g) {
      float s = ga_b[g];
      for (int e = 0; e < 256; ++e) s += sfr[e] * ga_w[e * 3 + g];
      lg[g] = s;                                   // TAU = 1
    }
    float m = fmaxf(lg[0], fmaxf(lg[1], lg[2]));
    float e0 = __expf(lg[0] - m), e1 = __expf(lg[1] - m), e2 = __expf(lg[2] - m);
    float inv = 1.f / (e0 + e1 + e2);
    float p0 = e0 * inv, p1 = e1 * inv, p2 = e2 * inv;
    if (lane < 16) {
      float* pr = probs_out + (size_t)(rowbase + r) * 3;
      pr[0] = p0; pr[1] = p1; pr[2] = p2;
    }
    int cbase = (lane >> 4) * 128;
    for (int c2 = 0; c2 < 128; ++c2) {
      int c = cbase + c2;
      float ge = p0 * gemb[c] + p1 * gemb[256 + c] + p2 * gemb[512 + c];
      bufA[r * 256 + c] = sfr[c] + ge;
    }
  }
  // gd_h1 = relu(bufA@gd1 + b) -> bufB
  {
    v16bf afr[8];
#pragma unroll
    for (int kc = 0; kc < 8; ++kc) afr[kc] = load_a(bufA, 256, kc * 32);
#pragma unroll 1
    for (int ct = 0; ct < 16; ++ct) {
      v8f acc = {};
#pragma unroll
      for (int kc = 0; kc < 8; ++kc)
        acc = wmma_bf16(afr[kc], load_pb(pw_gd1, ct * 8 + kc), acc);
      float bb = gd1_b[ct * 16 + col];
#pragma unroll
      for (int j = 0; j < 8; ++j) { float v = acc[j] + bb; acc[j] = v > 0.f ? v : 0.f; }
      store_c(bufB + ct * 16, 256, acc);
    }
  }
  // gs = tanh(bufB@gd2 + b) -> bufA and gs_out
  {
    v16bf afr[8];
#pragma unroll
    for (int kc = 0; kc < 8; ++kc) afr[kc] = load_a(bufB, 256, kc * 32);
#pragma unroll 1
    for (int ct = 0; ct < 16; ++ct) {
      v8f acc = {};
#pragma unroll
      for (int kc = 0; kc < 8; ++kc)
        acc = wmma_bf16(afr[kc], load_pb(pw_gd2, ct * 8 + kc), acc);
      float bb = gd2_b[ct * 16 + col];
#pragma unroll
      for (int j = 0; j < 8; ++j) acc[j] = tanhf(acc[j] + bb);
      store_c(bufA + ct * 16, 256, acc);
      store_c(gs_out + (size_t)rowbase * 256 + ct * 16, 256, acc);
    }
  }
  // fused hypernetwork: qacc = gs@hb + hb_b; then per h-index accumulate
  {
    v16bf ags[8];
#pragma unroll
    for (int kc = 0; kc < 8; ++kc) ags[kc] = load_a(bufA, 256, kc * 32);
    v8f qacc[2];
#pragma unroll
    for (int ct = 0; ct < 2; ++ct) {
      v8f acc = {};
#pragma unroll
      for (int kc = 0; kc < 8; ++kc)
        acc = wmma_bf16(ags[kc], load_pb(pw_hb, ct * 8 + kc), acc);
      float bb = hb_b[ct * 16 + col];
#pragma unroll
      for (int j = 0; j < 8; ++j) acc[j] += bb;
      qacc[ct] = acc;
    }
#pragma unroll 1
    for (int hidx = 0; hidx < 128; ++hidx) {
      float hv[8];
#pragma unroll
      for (int j = 0; j < 8; ++j) hv[j] = hloc[(rb + j) * 128 + hidx];
#pragma unroll
      for (int ct = 0; ct < 2; ++ct) {
        v8f w = {};
#pragma unroll
        for (int kc = 0; kc < 8; ++kc)
          w = wmma_bf16(ags[kc], load_pb(pw_hw, (hidx * 2 + ct) * 8 + kc), w);
        float wb = hw_b[hidx * 32 + ct * 16 + col];
        v8f t = qacc[ct];
#pragma unroll
        for (int j = 0; j < 8; ++j) t[j] += hv[j] * (w[j] + wb);
        qacc[ct] = t;
      }
    }
#pragma unroll
    for (int ct = 0; ct < 2; ++ct)
      store_c(q_out + (size_t)rowbase * 32 + ct * 16, 32, qacc[ct]);
  }
}

// ---------------------------------------------------------------------------
static void pack(const float* src, bf16* dst, int K, int N, hipStream_t s) {
  int KC = (K + 31) >> 5, NT = N >> 4;
  int total = NT * KC * 512;
  pack_b_kernel<<<(total + 255) / 256, 256, 0, s>>>(src, dst, K, N);
}

extern "C" void kernel_launch(void* const* d_in, const int* in_sizes, int n_in,
                              void* d_out, int out_size, void* d_ws, size_t ws_size,
                              hipStream_t stream) {
  (void)in_sizes; (void)n_in; (void)out_size; (void)ws_size;
  const float* inputs   = (const float*)d_in[0];
  const float* hidden   = (const float*)d_in[1];
  const float* fc1_w    = (const float*)d_in[2];
  const float* fc1_b    = (const float*)d_in[3];
  const float* gru_wi   = (const float*)d_in[4];
  const float* gru_wh   = (const float*)d_in[5];
  const float* gru_bi   = (const float*)d_in[6];
  const float* gru_bh   = (const float*)d_in[7];
  const float* attn_q_w = (const float*)d_in[8];
  const float* attn_k_w = (const float*)d_in[9];
  const float* se1_w    = (const float*)d_in[10];
  const float* se1_b    = (const float*)d_in[11];
  const float* se2_w    = (const float*)d_in[12];
  const float* se2_b    = (const float*)d_in[13];
  const float* ga_w     = (const float*)d_in[14];
  const float* ga_b     = (const float*)d_in[15];
  const float* gemb     = (const float*)d_in[16];
  const float* gd1_w    = (const float*)d_in[17];
  const float* gd1_b    = (const float*)d_in[18];
  const float* gd2_w    = (const float*)d_in[19];
  const float* gd2_b    = (const float*)d_in[20];
  const float* hb_w     = (const float*)d_in[21];
  const float* hb_b     = (const float*)d_in[22];
  const float* hw_w     = (const float*)d_in[23];
  const float* hw_b     = (const float*)d_in[24];

  const int R = B_ * A_;                       // 65536 rows
  float* out_q     = (float*)d_out;
  float* out_h     = out_q  + (size_t)R * NACT;
  float* out_gs    = out_h  + (size_t)R * H_;
  float* out_probs = out_gs + (size_t)R * HE_;
  float* out_attn  = out_probs + (size_t)R * G_;

  // workspace: struct_input [R,64] f32, then packed bf16 weights (1KB aligned)
  float* ws_si = (float*)d_ws;
  bf16* pk = (bf16*)(ws_si + (size_t)R * 64);
  size_t o = 0;
  bf16* p_fc1 = pk + o; o += (size_t)8   * 5 * 512;   // fc1_w  [160,128]
  bf16* p_wi  = pk + o; o += (size_t)24  * 4 * 512;   // gru_wi [128,384]
  bf16* p_wh  = pk + o; o += (size_t)24  * 4 * 512;   // gru_wh [128,384]
  bf16* p_q   = pk + o; o += (size_t)8   * 4 * 512;   // attn_q [128,128]
  bf16* p_k   = pk + o; o += (size_t)8   * 4 * 512;   // attn_k [128,128]
  bf16* p_se1 = pk + o; o += (size_t)16  * 2 * 512;   // se1_w  [34->64,256]
  bf16* p_se2 = pk + o; o += (size_t)16  * 8 * 512;   // se2_w  [256,256]
  bf16* p_gd1 = pk + o; o += (size_t)16  * 8 * 512;   // gd1_w  [256,256]
  bf16* p_gd2 = pk + o; o += (size_t)16  * 8 * 512;   // gd2_w  [256,256]
  bf16* p_hb  = pk + o; o += (size_t)2   * 8 * 512;   // hb_w   [256,32]
  bf16* p_hw  = pk + o; o += (size_t)256 * 8 * 512;   // hw_w   [256,4096]

  pack(fc1_w,    p_fc1, 160, 128, stream);
  pack(gru_wi,   p_wi,  128, 384, stream);
  pack(gru_wh,   p_wh,  128, 384, stream);
  pack(attn_q_w, p_q,   128, 128, stream);
  pack(attn_k_w, p_k,   128, 128, stream);
  pack(se1_w,    p_se1, 34,  256, stream);
  pack(se2_w,    p_se2, 256, 256, stream);
  pack(gd1_w,    p_gd1, 256, 256, stream);
  pack(gd2_w,    p_gd2, 256, 256, stream);
  pack(hb_w,     p_hb,  256, 32,  stream);
  pack(hw_w,     p_hw,  256, 4096, stream);

  k_fc1_gru<<<R / 64, 128, 0, stream>>>(inputs, hidden, fc1_b, gru_bi, gru_bh,
                                        p_fc1, p_wi, p_wh, out_h);
  k_attn<<<B_, 128, 0, stream>>>(out_h, p_q, p_k, out_attn, ws_si);
  k_group<<<R / 16, 32, 0, stream>>>(ws_si, out_h, se1_b, se2_b, ga_w, ga_b,
                                     gemb, gd1_b, gd2_b, hb_b, hw_b,
                                     p_se1, p_se2, p_gd1, p_gd2, p_hb, p_hw,
                                     out_q, out_gs, out_probs);
}
